// DKNet_51548197487210
// MI455X (gfx1250) — matrix operand order repaired and checked by reading, not compiled, not run on hardware
//
#include <hip/hip_runtime.h>
#include <hip/hip_bf16.h>

// ---------------------------------------------------------------------------
// CDNA5 (gfx1250) implementation. All GEMMs go through v_wmma_f32_16x16x32_f16
// (wave32 WMMA). f16 operands, f32 accumulate. BN layers: GEMM + atomic column
// stats; normalization+relu fused into the consumer's A-tile load.
// ---------------------------------------------------------------------------

typedef __attribute__((ext_vector_type(16))) _Float16 v16h;
typedef __attribute__((ext_vector_type(8)))  float    v8f;

#define NPTS 131072
#define KCAND 64
#define PSAMP 64
#define NB 4
#define NEGINF (-1e30f)

// ---------------------------------------------------------------------------
// Generic fused MLP layer (templated on K-padding and #16-col output tiles so
// every fragment loop fully unrolls -> no v_movrels, immediate LDS offsets):
//   Z = act(X) @ W (+ bias),  act = relu(bn(.)) if stats0 given, else identity
//   X = [X0 (d0 cols, optionally normalized) | X1 (d1 cols, raw)]
//   optionally accumulates column sum/sumsq of Z into stats_out (len 96:
//   [0..47]=sum, [48..95]=sumsq).
// Block: 256 threads = 8 waves, 128 rows.
// ---------------------------------------------------------------------------
template<int KP, int NT>
__global__ __launch_bounds__(256) void mlp_layer_kernel(
    const float* __restrict__ X0, int d0, const float* __restrict__ stats0, float invN0,
    const float* __restrict__ X1, int d1,
    const float* __restrict__ W, const float* __restrict__ bias,
    int d_out, int N,
    float* __restrict__ Z, float* __restrict__ stats_out)
{
    __shared__ _Float16 Ast[128][KP + 8];   // row-major A tile (f16), padded stride
    __shared__ _Float16 Bt[NT * 16][KP];    // transposed W: Bt[n][k]
    __shared__ float    Bs[NT * 16];

    const int tid  = threadIdx.x;
    const int d_in = d0 + d1;
    const long rowBase = (long)blockIdx.x * 128;

    // ---- stage A (apply bn+relu of producer layer on the fly) ----
    for (int idx = tid; idx < 128 * KP; idx += 256) {
        int r = idx / KP, c = idx - r * KP;
        long row = rowBase + r;
        float v = 0.f;
        if (row < N) {
            if (c < d0) {
                v = X0[row * d0 + c];
                if (stats0) {
                    float m   = stats0[c] * invN0;
                    float var = stats0[48 + c] * invN0 - m * m;
                    v = (v - m) * rsqrtf(var + 1e-5f);
                    v = fmaxf(v, 0.f);
                }
            } else if (c < d_in) {
                v = X1[row * d1 + (c - d0)];
            }
        }
        Ast[r][c] = (_Float16)v;
    }
    // ---- stage B transposed, zero padded ----
    for (int idx = tid; idx < NT * 16 * KP; idx += 256) {
        int n = idx / KP, c = idx - n * KP;
        float v = (n < d_out && c < d_in) ? W[c * d_out + n] : 0.f;
        Bt[n][c] = (_Float16)v;
    }
    for (int n = tid; n < NT * 16; n += 256)
        Bs[n] = (bias && n < d_out) ? bias[n] : 0.f;
    __syncthreads();

    const int lane = tid & 31, wave = tid >> 5;
    const int ml = lane & 15, hi = lane >> 4;
    const int arow = wave * 16 + ml;

    // A fragments per ISA 7.12.2: elem i<8  -> K = kt*32 + 8*hi + i
    //                             elem i>=8 -> K = kt*32 + 16 + 8*hi + (i-8)
    constexpr int NKT = KP / 32;
    v16h afrag[NKT];
#pragma unroll
    for (int kt = 0; kt < NKT; kt++) {
        v16h a;
        const _Float16* pa = &Ast[arow][kt * 32];
#pragma unroll
        for (int i = 0; i < 8; i++) a[i]     = pa[8 * hi + i];
#pragma unroll
        for (int i = 0; i < 8; i++) a[8 + i] = pa[16 + 8 * hi + i];
        afrag[kt] = a;
    }

#pragma unroll
    for (int nt = 0; nt < NT; nt++) {
        v8f acc = {};
#pragma unroll
        for (int kt = 0; kt < NKT; kt++) {
            // B fragment: elem i<8 -> K = kt*32 + 8*hi + i, elem i>=8 -> K = kt*32 + 16 + 8*hi + (i-8), col n = ml
            v16h b;
            const _Float16* pb = &Bt[nt * 16 + ml][kt * 32 + 8 * hi];
#pragma unroll
            for (int i = 0; i < 16; i++) b[i] = pb[i < 8 ? i : i + 8];
            acc = __builtin_amdgcn_wmma_f32_16x16x32_f16(
                false, afrag[kt], false, b, (short)0, acc, false, false);
        }
        const int col = nt * 16 + ml;
        const bool colok = (col < d_out);
        const float bv = Bs[nt * 16 + ml];
        float s1 = 0.f, s2 = 0.f;
#pragma unroll
        for (int r = 0; r < 8; r++) {
            long row = rowBase + wave * 16 + r + 8 * hi;   // D row = r + 8*hi
            float v = acc[r] + bv;
            bool ok = colok && (row < (long)N);
            float vm = ok ? v : 0.f;
            s1 += vm; s2 += vm * vm;
            if (ok) Z[row * d_out + col] = v;
        }
        if (stats_out) {
            s1 += __shfl_down(s1, 16, 32);   // lane l and l+16 share a column
            s2 += __shfl_down(s2, 16, 32);
            if (hi == 0 && colok) {
                atomicAdd(&stats_out[col],      s1);
                atomicAdd(&stats_out[48 + col], s2);
            }
        }
    }
}

// ---------------------------------------------------------------------------
// Mask head: for each candidate k:
//   h = relu([f_masks | xyz-center_k] @ w1[k] + b1[k]);  out = h @ w2[k] + b2[k]
// Block stages a 128-row A tile ONCE and loops all 64 candidates over it
// (f_masks occupies A-frag elems 0..7; pos columns built in registers into
//  elems 8..10 of the hi=0 half). The 16->1 contraction h@w2 is a SECOND WMMA
// with w2 in column 0 of B: h goes through a per-wave LDS tile (stride 24
// halves -> the row-major b128 reloads hit all 64 LDS banks exactly once),
// avoiding the 32-bpermute butterfly per candidate.
// ---------------------------------------------------------------------------
__global__ __launch_bounds__(256) void mask_head_kernel(
    const float* __restrict__ f_masks, const float* __restrict__ xyz,
    const float* __restrict__ wg, const int* __restrict__ batch_idx,
    const int* __restrict__ cand_batch, const float* __restrict__ centers,
    float* __restrict__ out, int N)
{
    __shared__ _Float16 Amst[128][16];
    __shared__ float    Xs[128][3];
    __shared__ _Float16 Bt[16][32];
    __shared__ _Float16 Htile[8][16][24];    // per-wave h round-trip (wave-private)
    __shared__ _Float16 W2h[16];
    __shared__ float B1s[16], Cent[3];
    __shared__ float B2s;
    __shared__ int   CB;

    const int tid = threadIdx.x;
    const long rowBase = (long)blockIdx.x * 128;

    for (int idx = tid; idx < 128 * 16; idx += 256) {
        int r = idx >> 4, c = idx & 15;
        Amst[r][c] = (_Float16)f_masks[(rowBase + r) * 16 + c];
    }
    for (int idx = tid; idx < 128 * 3; idx += 256) {
        int r = idx / 3, c = idx - r * 3;
        Xs[r][c] = xyz[(rowBase + r) * 3 + c];
    }
    __syncthreads();

    const int lane = tid & 31, wave = tid >> 5;
    const int ml = lane & 15, hi = lane >> 4;
    const int rloc = wave * 16 + ml;

    _Float16 am[8];
#pragma unroll
    for (int i = 0; i < 8; i++) am[i] = Amst[rloc][8 * hi + i];  // K = 8*hi+i (cols 0..15)
    const float px = Xs[rloc][0], py = Xs[rloc][1], pz = Xs[rloc][2];

    int br[8];
    if (ml == 0) {
#pragma unroll
        for (int r = 0; r < 8; r++)
            br[r] = batch_idx[rowBase + wave * 16 + r + 8 * hi];
    }

    for (int k = 0; k < KCAND; k++) {
        __syncthreads();                         // prior iter done with Bt/W2h
        const float* wgk = wg + k * 337;
        // pull next candidate's generated weights toward the caches
        // (lowers to global_prefetch_b8 on gfx1250)
        if (k + 1 < KCAND) __builtin_prefetch(wgk + 337, 0, 1);
        for (int idx = tid; idx < 16 * 32; idx += 256) {
            int n = idx >> 5, c = idx & 31;      // w1 = wg[:,:304].reshape(19,16)
            Bt[n][c] = (_Float16)((c < 19) ? wgk[c * 16 + n] : 0.f);
        }
        if (tid < 16) { B1s[tid] = wgk[304 + tid]; W2h[tid] = (_Float16)wgk[320 + tid]; }
        if (tid == 0) {
            B2s = wgk[336]; CB = cand_batch[k];
            Cent[0] = centers[k * 3 + 0]; Cent[1] = centers[k * 3 + 1]; Cent[2] = centers[k * 3 + 2];
        }
        __syncthreads();

        const _Float16 z0 = (_Float16)0.f;
        v16h a;
#pragma unroll
        for (int i = 0; i < 8; i++) a[i] = am[i];
        // elems 8..15: hi=0 -> K=16..23 (16,17,18 = pos); hi=1 -> K=24..31 (pad)
        a[8]  = (hi == 0) ? (_Float16)(px - Cent[0]) : z0;
        a[9]  = (hi == 0) ? (_Float16)(py - Cent[1]) : z0;
        a[10] = (hi == 0) ? (_Float16)(pz - Cent[2]) : z0;
        a[11] = z0; a[12] = z0; a[13] = z0; a[14] = z0; a[15] = z0;

        v16h b;
#pragma unroll
        for (int i = 0; i < 16; i++) b[i] = Bt[ml][i < 8 ? 8 * hi + i : 8 + 8 * hi + i];

        v8f acc = {};
        acc = __builtin_amdgcn_wmma_f32_16x16x32_f16(
            false, a, false, b, (short)0, acc, false, false);

        // h = relu(acc + b1): lane holds column ml, rows r+8*hi. Store to the
        // wave-private LDS tile to transpose into A-layout for WMMA #2.
        const float b1 = B1s[ml];
#pragma unroll
        for (int r = 0; r < 8; r++)
            Htile[wave][r + 8 * hi][ml] = (_Float16)fmaxf(acc[r] + b1, 0.f);

        // WMMA #2: out = h @ w2col.  A row m=ml: elems 0..7 = h[ml][8*hi..],
        // elems 8..15 = K-pad zeros.  B nonzero only for lanes 0/16 (n=0, K=8*hi+i, i<8).
        v16h a2;
        const _Float16* ph = &Htile[wave][ml][8 * hi];
#pragma unroll
        for (int i = 0; i < 8; i++) a2[i] = ph[i];
#pragma unroll
        for (int i = 8; i < 16; i++) a2[i] = z0;

        v16h b2;
#pragma unroll
        for (int i = 0; i < 16; i++) b2[i] = z0;
        if (ml == 0) {
#pragma unroll
            for (int i = 0; i < 8; i++) b2[i] = W2h[8 * hi + i];
        }

        v8f acc2 = {};
        acc2 = __builtin_amdgcn_wmma_f32_16x16x32_f16(
            false, a2, false, b2, (short)0, acc2, false, false);

        if (ml == 0) {   // lanes 0 and 16 hold D2 column 0 (rows r+8*hi)
#pragma unroll
            for (int r = 0; r < 8; r++) {
                long row = rowBase + wave * 16 + r + 8 * hi;
                float o = acc2[r] + B2s;
                o = (br[r] == CB) ? o : 0.f;
                out[(long)k * N + row] = o;
            }
        }
    }
}

// ------------------------- small helper kernels ----------------------------

__global__ __launch_bounds__(256) void init_kernel(float* stats, float* den) {
    int t = blockIdx.x * 256 + threadIdx.x;
    if (t < 16 * 96) stats[t] = 0.f;
    if (t < NB)      den[t]   = 0.f;
}

__global__ __launch_bounds__(256) void seg_max_part_kernel(
    const float* __restrict__ s, const int* __restrict__ b, int N, float* part) {
    __shared__ float sm[256 * 4];
    float m[4] = {NEGINF, NEGINF, NEGINF, NEGINF};
    for (int i = blockIdx.x * 256 + threadIdx.x; i < N; i += gridDim.x * 256) {
        float v = s[i]; int bb = b[i];
#pragma unroll
        for (int c = 0; c < 4; c++) if (bb == c) m[c] = fmaxf(m[c], v);
    }
#pragma unroll
    for (int c = 0; c < 4; c++) sm[threadIdx.x * 4 + c] = m[c];
    __syncthreads();
    for (int off = 128; off > 0; off >>= 1) {
        if (threadIdx.x < off)
#pragma unroll
            for (int c = 0; c < 4; c++)
                sm[threadIdx.x * 4 + c] = fmaxf(sm[threadIdx.x * 4 + c], sm[(threadIdx.x + off) * 4 + c]);
        __syncthreads();
    }
    if (threadIdx.x < 4) part[blockIdx.x * 4 + threadIdx.x] = sm[threadIdx.x];
}

__global__ __launch_bounds__(256) void seg_max_fin_kernel(const float* part, int nb, float* mx) {
    __shared__ float sm[256 * 4];
    float m[4] = {NEGINF, NEGINF, NEGINF, NEGINF};
    for (int i = threadIdx.x; i < nb; i += 256)
#pragma unroll
        for (int c = 0; c < 4; c++) m[c] = fmaxf(m[c], part[i * 4 + c]);
#pragma unroll
    for (int c = 0; c < 4; c++) sm[threadIdx.x * 4 + c] = m[c];
    __syncthreads();
    for (int off = 128; off > 0; off >>= 1) {
        if (threadIdx.x < off)
#pragma unroll
            for (int c = 0; c < 4; c++)
                sm[threadIdx.x * 4 + c] = fmaxf(sm[threadIdx.x * 4 + c], sm[(threadIdx.x + off) * 4 + c]);
        __syncthreads();
    }
    if (threadIdx.x < 4) mx[threadIdx.x] = sm[threadIdx.x];
}

__global__ __launch_bounds__(256) void seg_exp_kernel(
    const float* __restrict__ s, const int* __restrict__ b, const float* __restrict__ mx,
    int N, float* __restrict__ e, float* den) {
    __shared__ float sm[256 * 4];
    float a[4] = {0.f, 0.f, 0.f, 0.f};
    for (int i = blockIdx.x * 256 + threadIdx.x; i < N; i += gridDim.x * 256) {
        int bb = b[i];
        float v = __expf(s[i] - mx[bb]);
        e[i] = v;
#pragma unroll
        for (int c = 0; c < 4; c++) if (bb == c) a[c] += v;
    }
#pragma unroll
    for (int c = 0; c < 4; c++) sm[threadIdx.x * 4 + c] = a[c];
    __syncthreads();
    for (int off = 128; off > 0; off >>= 1) {
        if (threadIdx.x < off)
#pragma unroll
            for (int c = 0; c < 4; c++)
                sm[threadIdx.x * 4 + c] += sm[(threadIdx.x + off) * 4 + c];
        __syncthreads();
    }
    if (threadIdx.x < 4) atomicAdd(&den[threadIdx.x], sm[threadIdx.x]);
}

__global__ __launch_bounds__(256) void seg_score_kernel(
    const float* __restrict__ e, const int* __restrict__ b, const float* __restrict__ den,
    int N, float* __restrict__ out) {
    for (int i = blockIdx.x * 256 + threadIdx.x; i < N; i += gridDim.x * 256)
        out[i] = e[i] / den[b[i]];
}

__global__ __launch_bounds__(128) void gather_mean_kernel(
    const float* __restrict__ fk, const int* __restrict__ idx, float* __restrict__ out) {
    __shared__ float red[8][16];
    int k = blockIdx.x, tid = threadIdx.x;
    int col = tid & 15, pg = tid >> 4;
    float s = 0.f;
    for (int p = pg; p < PSAMP; p += 8)
        s += fk[(long)idx[k * PSAMP + p] * 16 + col];
    red[pg][col] = s;
    __syncthreads();
    if (pg == 0) {
        float t = 0.f;
#pragma unroll
        for (int j = 0; j < 8; j++) t += red[j][col];
        out[k * 16 + col] = t * (1.f / (float)PSAMP);
    }
}

__global__ __launch_bounds__(256) void fcand_kernel(
    const float* __restrict__ ffg, const float* __restrict__ fbg, const float* __restrict__ xyz,
    const float* __restrict__ poff, const int* __restrict__ cand, const int* __restrict__ bidx,
    float* __restrict__ fc, float* __restrict__ centers, int* __restrict__ cb) {
    int tid = threadIdx.x;
    for (int idx = tid; idx < KCAND * 35; idx += 256) {
        int k = idx / 35, c = idx - 35 * k;
        float v;
        if (c < 16)      v = ffg[k * 16 + c];
        else if (c < 32) v = fbg[k * 16 + (c - 16)];
        else {
            int cp = cand[k]; int cc = c - 32;
            v = xyz[(long)cp * 3 + cc] + poff[(long)cp * 3 + cc];
        }
        fc[idx] = v;
    }
    if (tid < KCAND) {
        int cp = cand[tid];
        cb[tid] = bidx[cp];
        centers[tid * 3 + 0] = xyz[(long)cp * 3 + 0];
        centers[tid * 3 + 1] = xyz[(long)cp * 3 + 1];
        centers[tid * 3 + 2] = xyz[(long)cp * 3 + 2];
    }
}

__global__ __launch_bounds__(256) void dmat_kernel(const float* __restrict__ fc, float* __restrict__ dm) {
    int idx = blockIdx.x * 256 + threadIdx.x;
    const int T = KCAND * KCAND * 35;
    if (idx >= T) return;
    int c = idx % 35; int ij = idx / 35; int i = ij >> 6, j = ij & 63;
    dm[idx] = fmaxf(fabsf(fc[i * 35 + c] - fc[j * 35 + c]), 1e-6f);
}

__global__ __launch_bounds__(256) void merge_kernel(
    const float* __restrict__ logit, const int* __restrict__ cb, float* __restrict__ out) {
    int t = blockIdx.x * 256 + threadIdx.x;
    if (t >= KCAND * KCAND) return;
    int i = t >> 6, j = t & 63;
    float v = 1.f / (1.f + __expf(-logit[t]));
    out[t] = (cb[i] == cb[j]) ? v : 0.f;
}

__global__ __launch_bounds__(256) void wg2_kernel(
    const float* __restrict__ wgh, const float* __restrict__ st,
    const float* __restrict__ w2, const float* __restrict__ b2, float* __restrict__ wg) {
    int idx = blockIdx.x * 256 + threadIdx.x;
    if (idx >= KCAND * 337) return;
    int k = idx / 337, j = idx - k * 337;
    const float invn = 1.f / (float)KCAND;
    float acc = b2[j];
#pragma unroll
    for (int c = 0; c < 16; c++) {
        float m = st[c] * invn;
        float var = st[48 + c] * invn - m * m;
        float h = fmaxf((wgh[k * 16 + c] - m) * rsqrtf(var + 1e-5f), 0.f);
        acc += h * w2[c * 337 + j];
    }
    wg[idx] = acc;
}

// ------------------------------- host side ---------------------------------

// stats slot ids (96 floats each)
enum { S_SEM1, S_SEM2, S_OFF1, S_HM1, S_HM2, S_KR1, S_KR2, S_KR3,
       S_MK1, S_MK2, S_MK3, S_AG1, S_AG2, S_AG3, S_WG1, S_NSLOTS };

static void launch_layer(hipStream_t st,
                         const float* X0, int d0, const float* stats0, float invN0,
                         const float* X1, int d1,
                         const float* W, const float* bias,
                         int d_out, int N, float* Z, float* stats_out) {
    const int d_in = d0 + d1;
    const int kp = (d_in <= 32) ? 32 : 64;
    const int nt = (d_out + 15) >> 4;
    dim3 g((unsigned)((N + 127) / 128));
    if (kp == 32 && nt == 1)
        mlp_layer_kernel<32, 1><<<g, 256, 0, st>>>(X0, d0, stats0, invN0, X1, d1, W, bias, d_out, N, Z, stats_out);
    else if (kp == 32 && nt == 2)
        mlp_layer_kernel<32, 2><<<g, 256, 0, st>>>(X0, d0, stats0, invN0, X1, d1, W, bias, d_out, N, Z, stats_out);
    else if (kp == 64 && nt == 1)
        mlp_layer_kernel<64, 1><<<g, 256, 0, st>>>(X0, d0, stats0, invN0, X1, d1, W, bias, d_out, N, Z, stats_out);
    else if (kp == 64 && nt == 2)
        mlp_layer_kernel<64, 2><<<g, 256, 0, st>>>(X0, d0, stats0, invN0, X1, d1, W, bias, d_out, N, Z, stats_out);
    else
        mlp_layer_kernel<64, 3><<<g, 256, 0, st>>>(X0, d0, stats0, invN0, X1, d1, W, bias, d_out, N, Z, stats_out);
}

extern "C" void kernel_launch(void* const* d_in, const int* in_sizes, int n_in,
                              void* d_out, int out_size, void* d_ws, size_t ws_size,
                              hipStream_t stream) {
    const int N = NPTS;
    const float invN  = 1.f / (float)N;
    const float invNA = 1.f / 4096.f;

    // inputs in setup_inputs() dict insertion order (params flattened in place)
    const float* f_3d  = (const float*)d_in[0];
    const float* xyz   = (const float*)d_in[1];
    const float* sem_w1 = (const float*)d_in[2];  const float* sem_b1 = (const float*)d_in[3];
    const float* sem_w2 = (const float*)d_in[4];  const float* sem_b2 = (const float*)d_in[5];
    const float* sem_w3 = (const float*)d_in[6];  const float* sem_b3 = (const float*)d_in[7];
    const float* off_w1 = (const float*)d_in[8];  const float* off_b1 = (const float*)d_in[9];
    const float* off_w2 = (const float*)d_in[10]; const float* off_b2 = (const float*)d_in[11];
    const float* hm_w1  = (const float*)d_in[12]; const float* hm_b1  = (const float*)d_in[13];
    const float* hm_w2  = (const float*)d_in[14]; const float* hm_b2  = (const float*)d_in[15];
    const float* hm_w3  = (const float*)d_in[16]; const float* hm_b3  = (const float*)d_in[17];
    const float* kr_w1  = (const float*)d_in[18]; const float* kr_w2  = (const float*)d_in[19];
    const float* kr_w3  = (const float*)d_in[20]; const float* kr_w4  = (const float*)d_in[21];
    const float* kr_b4  = (const float*)d_in[22];
    const float* ag_w1  = (const float*)d_in[23]; const float* ag_w2  = (const float*)d_in[24];
    const float* ag_w3  = (const float*)d_in[25]; const float* ag_w4  = (const float*)d_in[26];
    const float* ag_b4  = (const float*)d_in[27];
    const float* mk_w1  = (const float*)d_in[28]; const float* mk_w2  = (const float*)d_in[29];
    const float* mk_w3  = (const float*)d_in[30]; const float* mk_w4  = (const float*)d_in[31];
    const float* mk_b4  = (const float*)d_in[32];
    const float* wg_w1  = (const float*)d_in[33]; const float* wg_w2  = (const float*)d_in[34];
    const float* wg_b2  = (const float*)d_in[35];
    const int* batch_idx = (const int*)d_in[36];
    const int* cand      = (const int*)d_in[37];
    const int* kfg       = (const int*)d_in[38];
    const int* kbg       = (const int*)d_in[39];

    // output layout (return order): mask(K*N), merge(K*K), sem(N*20), score(N)
    float* out_mask  = (float*)d_out;
    float* out_merge = out_mask + (long)KCAND * N;
    float* out_sem   = out_merge + KCAND * KCAND;
    float* out_score = out_sem + (long)N * 20;

    // workspace layout (floats)
    float* ws = (float*)d_ws;
    long o = 0;
    float* ZA    = ws + o; o += (long)N * 32;
    float* ZB    = ws + o; o += (long)N * 32;
    float* OFF   = ws + o; o += (long)N * 3;
    float* SHEAT = ws + o; o += N;
    float* EEXP  = ws + o; o += N;
    float* FK    = ws + o; o += (long)N * 16;
    float* FM    = ws + o; o += (long)N * 16;
    float* FFG   = ws + o; o += KCAND * 16;
    float* FBG   = ws + o; o += KCAND * 16;
    float* FCAND = ws + o; o += KCAND * 35;
    float* CENT  = ws + o; o += KCAND * 3;
    int*   CB    = (int*)(ws + o); o += KCAND;
    float* WGH   = ws + o; o += KCAND * 16;
    float* WGBUF = ws + o; o += KCAND * 337;
    float* LOGIT = ws + o; o += KCAND * KCAND;
    float* STATS = ws + o; o += S_NSLOTS * 96 + 32;
    float* MX    = ws + o; o += 4;
    float* DEN   = ws + o; o += 4;
    float* PART  = ws + o; o += 512 * 4;
    (void)ws_size; (void)in_sizes; (void)n_in; (void)out_size;

    auto st = [&](int slot) { return STATS + slot * 96; };

    init_kernel<<<6, 256, 0, stream>>>(STATS, DEN);

    // --- semantic branch ---
    launch_layer(stream, f_3d, 32, nullptr, 0.f, nullptr, 0, sem_w1, sem_b1, 32, N, ZA, st(S_SEM1));
    launch_layer(stream, ZA, 32, st(S_SEM1), invN, nullptr, 0, sem_w2, sem_b2, 32, N, ZB, st(S_SEM2));
    launch_layer(stream, ZB, 32, st(S_SEM2), invN, nullptr, 0, sem_w3, sem_b3, 20, N, out_sem, nullptr);
    // --- offset branch ---
    launch_layer(stream, f_3d, 32, nullptr, 0.f, nullptr, 0, off_w1, off_b1, 32, N, ZA, st(S_OFF1));
    launch_layer(stream, ZA, 32, st(S_OFF1), invN, nullptr, 0, off_w2, off_b2, 3, N, OFF, nullptr);
    // --- heatmap branch (input = [f_3d | point_offset]) ---
    launch_layer(stream, f_3d, 32, nullptr, 0.f, OFF, 3, hm_w1, hm_b1, 32, N, ZA, st(S_HM1));
    launch_layer(stream, ZA, 32, st(S_HM1), invN, nullptr, 0, hm_w2, hm_b2, 32, N, ZB, st(S_HM2));
    launch_layer(stream, ZB, 32, st(S_HM2), invN, nullptr, 0, hm_w3, hm_b3, 1, N, SHEAT, nullptr);
    // --- kernel branch ---
    launch_layer(stream, f_3d, 32, nullptr, 0.f, nullptr, 0, kr_w1, nullptr, 32, N, ZA, st(S_KR1));
    launch_layer(stream, ZA, 32, st(S_KR1), invN, nullptr, 0, kr_w2, nullptr, 32, N, ZB, st(S_KR2));
    launch_layer(stream, ZB, 32, st(S_KR2), invN, nullptr, 0, kr_w3, nullptr, 32, N, ZA, st(S_KR3));
    launch_layer(stream, ZA, 32, st(S_KR3), invN, nullptr, 0, kr_w4, kr_b4, 16, N, FK, nullptr);
    // --- mask-feature branch ---
    launch_layer(stream, f_3d, 32, nullptr, 0.f, nullptr, 0, mk_w1, nullptr, 32, N, ZA, st(S_MK1));
    launch_layer(stream, ZA, 32, st(S_MK1), invN, nullptr, 0, mk_w2, nullptr, 32, N, ZB, st(S_MK2));
    launch_layer(stream, ZB, 32, st(S_MK2), invN, nullptr, 0, mk_w3, nullptr, 32, N, ZA, st(S_MK3));
    launch_layer(stream, ZA, 32, st(S_MK3), invN, nullptr, 0, mk_w4, mk_b4, 16, N, FM, nullptr);

    // --- segment softmax over batch ---
    seg_max_part_kernel<<<512, 256, 0, stream>>>(SHEAT, batch_idx, N, PART);
    seg_max_fin_kernel<<<1, 256, 0, stream>>>(PART, 512, MX);
    seg_exp_kernel<<<512, 256, 0, stream>>>(SHEAT, batch_idx, MX, N, EEXP, DEN);
    seg_score_kernel<<<512, 256, 0, stream>>>(EEXP, batch_idx, DEN, N, out_score);

    // --- candidate features ---
    gather_mean_kernel<<<KCAND, 128, 0, stream>>>(FK, kfg, FFG);
    gather_mean_kernel<<<KCAND, 128, 0, stream>>>(FK, kbg, FBG);
    fcand_kernel<<<1, 256, 0, stream>>>(FFG, FBG, xyz, OFF, cand, batch_idx, FCAND, CENT, CB);

    // --- pairwise affinity MLP (N=4096 rows of width 35) ---
    dmat_kernel<<<(KCAND * KCAND * 35 + 255) / 256, 256, 0, stream>>>(FCAND, ZB);
    launch_layer(stream, ZB, 35, nullptr, 0.f, nullptr, 0, ag_w1, nullptr, 35, 4096, ZA, st(S_AG1));
    launch_layer(stream, ZA, 35, st(S_AG1), invNA, nullptr, 0, ag_w2, nullptr, 35, 4096, ZB, st(S_AG2));
    launch_layer(stream, ZB, 35, st(S_AG2), invNA, nullptr, 0, ag_w3, nullptr, 35, 4096, ZA, st(S_AG3));
    launch_layer(stream, ZA, 35, st(S_AG3), invNA, nullptr, 0, ag_w4, ag_b4, 1, 4096, LOGIT, nullptr);
    merge_kernel<<<(KCAND * KCAND + 255) / 256, 256, 0, stream>>>(LOGIT, CB, out_merge);

    // --- dynamic-weight generation ---
    launch_layer(stream, FFG, 16, nullptr, 0.f, nullptr, 0, wg_w1, nullptr, 16, KCAND, WGH, st(S_WG1));
    wg2_kernel<<<(KCAND * 337 + 255) / 256, 256, 0, stream>>>(WGH, st(S_WG1), wg_w2, wg_b2, WGBUF);

    // --- dynamic-filter mask head (dominant GEMM, WMMA) ---
    mask_head_kernel<<<N / 128, 256, 0, stream>>>(FM, xyz, WGBUF, batch_idx, CB, CENT, out_mask, N);
}
